// MultiHeadAttention_88218628260188
// MI455X (gfx1250) — matrix operand order repaired and checked by reading, not compiled, not run on hardware
//
#include <hip/hip_runtime.h>
#include <hip/hip_bf16.h>

typedef __attribute__((ext_vector_type(16))) __bf16 v16bf;
typedef __attribute__((ext_vector_type(8)))  __bf16 v8bf;
typedef __attribute__((ext_vector_type(8)))  float  v8f;

#define EMBED 1024
#define NHEAD 16
#define HDIM  64
#define TLEN  1024
#define BSZ   4
#define NROW  (TLEN * BSZ)    // 4096 token rows (t*B + b)
#define NBH   (BSZ * NHEAD)   // 64

// ---- WMMA helpers ------------------------------------------------------
__device__ __forceinline__ v16bf ld16bf(const __bf16* p) {
  // contiguous 32B: B-fragment per-lane chunk (K = half*16 .. +16)
  v8bf lo = *(const v8bf*)p;
  v8bf hi = *(const v8bf*)(p + 8);
  return __builtin_shufflevector(lo, hi, 0,1,2,3,4,5,6,7,8,9,10,11,12,13,14,15);
}

__device__ __forceinline__ v16bf ldAfrag(const __bf16* row, int kc, int half) {
  // A-fragment per-lane: two 16B chunks at K = kc+half*16 and kc+half*16+8
  v8bf lo = *(const v8bf*)(row + kc + half * 16);
  v8bf hi = *(const v8bf*)(row + kc + half * 16 + 8);
  return __builtin_shufflevector(lo, hi, 0,1,2,3,4,5,6,7,8,9,10,11,12,13,14,15);
}

__device__ __forceinline__ v8f wmma_bf16(v16bf a, v16bf b, v8f c) {
  return __builtin_amdgcn_wmma_f32_16x16x32_bf16(false, a, false, b, (short)0, c,
                                                 false, false);
}

// ---- fp32 -> bf16 convert ---------------------------------------------
__global__ void cvt_f32_bf16(const float* __restrict__ src,
                             __bf16* __restrict__ dst, int n) {
  int i = blockIdx.x * blockDim.x + threadIdx.x;
  if (i < n) dst[i] = (__bf16)src[i];
}

// ---- fused QKV projection: Y = X @ W^T + b, bf16 WMMA, fp32 accum ------
// outputs: Q,K in [BH][T][HDIM] bf16 (Q pre-scaled by Dh^-0.5), V in
// transposed [BH][HDIM][T] bf16 so P@V B-frags are contiguous.
__global__ void __launch_bounds__(128)
proj_qkv(const __bf16* __restrict__ xbf, const __bf16* __restrict__ wbf,
         const float* __restrict__ bq, const float* __restrict__ bk,
         const float* __restrict__ bv,
         __bf16* __restrict__ qout, __bf16* __restrict__ kout,
         __bf16* __restrict__ vTout) {
  const int lane = threadIdx.x, n = lane & 15, half = lane >> 4;
  const int p = blockIdx.z;                                // 0=Q 1=K 2=V
  const int r0 = (blockIdx.x * 4 + threadIdx.y) * 16;      // token-row tile
  const int c0 = blockIdx.y * 64;                          // 64 cols = 1 head
  const __bf16* W = wbf + (size_t)p * EMBED * EMBED;
  const float* bias = (p == 0) ? bq : (p == 1) ? bk : bv;
  const __bf16* arow = xbf + (size_t)(r0 + n) * EMBED;

  v8f acc[4] = {};
  for (int kc = 0; kc < EMBED; kc += 32) {
    v16bf a = ldAfrag(arow, kc, half);
#pragma unroll
    for (int j = 0; j < 4; ++j) {
      const __bf16* wr = W + (size_t)(c0 + j * 16 + n) * EMBED + kc + half * 16;
      acc[j] = wmma_bf16(a, ld16bf(wr), acc[j]);
    }
  }
#pragma unroll
  for (int j = 0; j < 4; ++j) {
    const int c = c0 + j * 16 + n;
    const int h = c >> 6, d = c & 63;
    const float bb = bias[c];
#pragma unroll
    for (int i = 0; i < 8; ++i) {
      const int r = r0 + i + 8 * half;       // flat token row = t*B + b
      const int t = r >> 2, b = r & 3;
      const int bh = b * NHEAD + h;
      float y = acc[j][i] + bb;
      if (p == 0)
        qout[((size_t)bh * TLEN + t) * HDIM + d] = (__bf16)(y * 0.125f);
      else if (p == 1)
        kout[((size_t)bh * TLEN + t) * HDIM + d] = (__bf16)y;
      else
        vTout[((size_t)bh * HDIM + d) * TLEN + t] = (__bf16)y;
    }
  }
}

// ---- fused attention: S=QK^T+bias, softmax, write head_weights, P@V ----
// 1 wave per block, 16-row strip of one (b,h); 64KB LDS score strip.
__global__ void __launch_bounds__(32)
attn_fused(const __bf16* __restrict__ qbf, const __bf16* __restrict__ kbf,
           const __bf16* __restrict__ vT, const float* __restrict__ bias,
           float* __restrict__ hw, __bf16* __restrict__ ctx) {
  __shared__ float sS[16 * TLEN];   // 64KB: this wave's 16x1024 score strip
  const int lane = threadIdx.x, n = lane & 15, half = lane >> 4;
  const int r0 = blockIdx.x * 16;
  const int bh = blockIdx.y;
  const int b = bh >> 4, h = bh & 15;

  // Q A-fragments for the whole strip (K-dim = HDIM = 2 chunks of 32)
  const __bf16* qrow = qbf + ((size_t)bh * TLEN + r0 + n) * HDIM;
  const v16bf a0 = ldAfrag(qrow, 0, half);
  const v16bf a1 = ldAfrag(qrow, 32, half);

  const float* biasbase = bias + ((size_t)bh * TLEN + r0) * TLEN;

  float rm[8];
#pragma unroll
  for (int i = 0; i < 8; ++i) rm[i] = -3.0e38f;

  // pass 1: scores + bias into LDS, running row max (C layout: row=i+8*half)
  for (int ct = 0; ct < 64; ++ct) {
    const __bf16* krow =
        kbf + ((size_t)bh * TLEN + ct * 16 + n) * HDIM + half * 16;
    if (ct < 63)
      __builtin_prefetch(kbf + ((size_t)bh * TLEN + (ct + 1) * 16 + n) * HDIM,
                         0, 0);
    v8f acc = {};
    acc = wmma_bf16(a0, ld16bf(krow), acc);
    acc = wmma_bf16(a1, ld16bf(krow + 32), acc);
#pragma unroll
    for (int i = 0; i < 8; ++i) {
      const int rl = i + 8 * half;
      float s = acc[i] + biasbase[(size_t)rl * TLEN + ct * 16 + n];
      sS[rl * TLEN + ct * 16 + n] = s;
      rm[i] = fmaxf(rm[i], s);
    }
  }
  __syncthreads();
#pragma unroll
  for (int i = 0; i < 8; ++i)
    for (int m = 1; m < 16; m <<= 1)
      rm[i] = fmaxf(rm[i], __shfl_xor(rm[i], m, 32));

  // pass 2: exp in place + row sums
  float rs[8] = {};
  for (int ct = 0; ct < 64; ++ct) {
#pragma unroll
    for (int i = 0; i < 8; ++i) {
      const int idx = (i + 8 * half) * TLEN + ct * 16 + n;
      float e = __expf(sS[idx] - rm[i]);
      sS[idx] = e;
      rs[i] += e;
    }
  }
  __syncthreads();
#pragma unroll
  for (int i = 0; i < 8; ++i)
    for (int m = 1; m < 16; m <<= 1) rs[i] += __shfl_xor(rs[i], m, 32);
  float inv[8];
#pragma unroll
  for (int i = 0; i < 8; ++i) inv[i] = 1.0f / rs[i];

  // pass 3a: normalized softmax -> head_weights [H,B,T,T] (single write)
  float* hwbase = hw + ((size_t)(h * BSZ + b) * TLEN + r0) * TLEN;
  for (int ct = 0; ct < 64; ++ct) {
#pragma unroll
    for (int i = 0; i < 8; ++i) {
      const int rl = i + 8 * half;
      hwbase[(size_t)rl * TLEN + ct * 16 + n] =
          sS[rl * TLEN + ct * 16 + n] * inv[i];
    }
  }

  // pass 3b: O = P @ V (unnormalized P from LDS; normalize in epilogue)
  v8f o[4] = {};
  const __bf16* vbase = vT + (size_t)bh * HDIM * TLEN;
  for (int kc = 0; kc < 32; ++kc) {
    const float* prow = sS + n * TLEN + kc * 32;   // A row = lane%16
    v8f f0 = *(const v8f*)(prow + half * 16);
    v8f f1 = *(const v8f*)(prow + half * 16 + 8);
    v16bf a;
#pragma unroll
    for (int e = 0; e < 8; ++e) {
      a[e] = (__bf16)f0[e];
      a[e + 8] = (__bf16)f1[e];
    }
#pragma unroll
    for (int j = 0; j < 4; ++j) {
      const __bf16* vrow =
          vbase + (size_t)(j * 16 + n) * TLEN + kc * 32 + half * 16;
      o[j] = wmma_bf16(a, ld16bf(vrow), o[j]);
    }
  }
  // epilogue: ctx[t][b][h*64+d] bf16 for the output projection
#pragma unroll
  for (int j = 0; j < 4; ++j)
#pragma unroll
    for (int i = 0; i < 8; ++i) {
      const int t = r0 + i + 8 * half;
      ctx[((size_t)t * BSZ + b) * EMBED + h * HDIM + j * 16 + n] =
          (__bf16)(o[j][i] * inv[i]);
    }
}

// ---- output projection: attn = ctx @ Wo^T + bo (fp32 out) --------------
__global__ void __launch_bounds__(128)
out_proj(const __bf16* __restrict__ ctx, const __bf16* __restrict__ wo,
         const float* __restrict__ bo, float* __restrict__ out) {
  const int lane = threadIdx.x, n = lane & 15, half = lane >> 4;
  const int r0 = (blockIdx.x * 4 + threadIdx.y) * 16;
  const int c0 = blockIdx.y * 64;
  const __bf16* arow = ctx + (size_t)(r0 + n) * EMBED;
  v8f acc[4] = {};
  for (int kc = 0; kc < EMBED; kc += 32) {
    v16bf a = ldAfrag(arow, kc, half);
#pragma unroll
    for (int j = 0; j < 4; ++j) {
      const __bf16* wr = wo + (size_t)(c0 + j * 16 + n) * EMBED + kc + half * 16;
      acc[j] = wmma_bf16(a, ld16bf(wr), acc[j]);
    }
  }
#pragma unroll
  for (int j = 0; j < 4; ++j) {
    const float bb = bo[c0 + j * 16 + n];
#pragma unroll
    for (int i = 0; i < 8; ++i) {
      const int r = r0 + i + 8 * half;
      out[(size_t)r * EMBED + c0 + j * 16 + n] = acc[j][i] + bb;
    }
  }
}

// ---- launcher ----------------------------------------------------------
extern "C" void kernel_launch(void* const* d_in, const int* in_sizes, int n_in,
                              void* d_out, int out_size, void* d_ws,
                              size_t ws_size, hipStream_t stream) {
  const float* query = (const float*)d_in[0];
  const float* attn_bias = (const float*)d_in[1];
  const float* wq = (const float*)d_in[2];
  const float* bq = (const float*)d_in[3];
  const float* wk = (const float*)d_in[4];
  const float* bk = (const float*)d_in[5];
  const float* wv = (const float*)d_in[6];
  const float* bv = (const float*)d_in[7];
  const float* wo = (const float*)d_in[8];
  const float* bo = (const float*)d_in[9];

  float* attn_out = (float*)d_out;                       // [T,B,E] fp32
  float* hw_out = attn_out + (size_t)NROW * EMBED;       // [H,B,T,T] fp32

  // workspace layout (bf16): xbf(8MB) | w q/k/v/o (8MB) | q,k,vT,ctx (8MB ea)
  __bf16* xbf = (__bf16*)d_ws;
  __bf16* wbf = xbf + (size_t)NROW * EMBED;
  __bf16* qbf = wbf + (size_t)4 * EMBED * EMBED;
  __bf16* kbf = qbf + (size_t)NBH * TLEN * HDIM;
  __bf16* vTb = kbf + (size_t)NBH * TLEN * HDIM;
  __bf16* ctx = vTb + (size_t)NBH * TLEN * HDIM;

  const int NT = 256;
  cvt_f32_bf16<<<(NROW * EMBED + NT - 1) / NT, NT, 0, stream>>>(query, xbf,
                                                                NROW * EMBED);
  cvt_f32_bf16<<<(EMBED * EMBED + NT - 1) / NT, NT, 0, stream>>>(
      wq, wbf + (size_t)0 * EMBED * EMBED, EMBED * EMBED);
  cvt_f32_bf16<<<(EMBED * EMBED + NT - 1) / NT, NT, 0, stream>>>(
      wk, wbf + (size_t)1 * EMBED * EMBED, EMBED * EMBED);
  cvt_f32_bf16<<<(EMBED * EMBED + NT - 1) / NT, NT, 0, stream>>>(
      wv, wbf + (size_t)2 * EMBED * EMBED, EMBED * EMBED);
  cvt_f32_bf16<<<(EMBED * EMBED + NT - 1) / NT, NT, 0, stream>>>(
      wo, wbf + (size_t)3 * EMBED * EMBED, EMBED * EMBED);

  proj_qkv<<<dim3(64, 16, 3), dim3(32, 4), 0, stream>>>(xbf, wbf, bq, bk, bv,
                                                        qbf, kbf, vTb);
  attn_fused<<<dim3(64, 64), dim3(32), 0, stream>>>(qbf, kbf, vTb, attn_bias,
                                                    hw_out, ctx);
  out_proj<<<dim3(64, 16), dim3(32, 4), 0, stream>>>(
      ctx, wbf + (size_t)3 * EMBED * EMBED, bo, attn_out);
}
